// CRFLoss_vb_549755814455
// MI455X (gfx1250) — matrix operand order repaired and checked by reading, not compiled, not run on hardware
//
#include <hip/hip_runtime.h>
#include <hip/hip_bf16.h>
#include <stdint.h>

// CRF forward loss for MI455X (gfx1250, wave32).
// scores: (L=256, B=128, T=64, T=64) f32; target: (L,B,1) i64; mask: (L,B) bool.
// Memory-bound: 536 MB stream -> ~23 us floor at 23.3 TB/s.
// One workgroup per batch (128 WGs), 4 waves each; per-step 16 KB tile is
// double-buffered in LDS via global_load_async_to_lds_b128; the log-semiring
// matvec runs on v_wmma_f32_16x16x32_f16 in the exp domain.

#define CRF_L 256
#define CRF_B 128
#define CRF_T 64
#define START_TAG 62
#define END_TAG 63

typedef __attribute__((ext_vector_type(16))) _Float16 v16h;
typedef __attribute__((ext_vector_type(8)))  float    v8f;

// ---------------------------------------------------------------- gather ----
__global__ __launch_bounds__(256)
void crf_gather_kernel(const float* __restrict__ scores,
                       const long long* __restrict__ target,
                       const unsigned char* __restrict__ mask,
                       float* __restrict__ ws_tg) {
  __shared__ float red[256];
  const int idx = blockIdx.x * 256 + threadIdx.x;   // 0 .. L*B-1 (=32768)
  const long long tgt = target[idx];                // 0 .. T*T-1
  float v = scores[(size_t)idx * (CRF_T * CRF_T) + (size_t)tgt];
  if (!mask[idx]) v = 0.0f;
  red[threadIdx.x] = v;
  __syncthreads();
  for (int s = 128; s > 0; s >>= 1) {
    if (threadIdx.x < s) red[threadIdx.x] += red[threadIdx.x + s];
    __syncthreads();
  }
  if (threadIdx.x == 0) ws_tg[blockIdx.x] = red[0];  // 128 partials
}

// --------------------------------------------------------------- forward ----
__global__ __launch_bounds__(128)
void crf_forward_kernel(const float* __restrict__ scores,
                        const unsigned char* __restrict__ mask,
                        float* __restrict__ ws_part) {
  __shared__ float tile[2][CRF_T * CRF_T];   // double-buffered 16 KB tiles
  __shared__ float part[CRF_T];
  __shared__ float pexp[CRF_T];

  const int b    = blockIdx.x;       // one workgroup per batch element
  const int t    = threadIdx.x;      // 0..127 (4 waves, wave32)
  const int wave = t >> 5;           // j-tile owner: columns [16w, 16w+16)
  const int lane = t & 31;
  const int hi16 = (lane >= 16) ? 1 : 0;
  const int j    = wave * 16 + (lane & 15);   // my output column

  // partition0 = scores[0, b, START_TAG, :]
  const float* s0 = scores + (size_t)b * (CRF_T * CRF_T);
  if (t < CRF_T) part[t] = s0[START_TAG * CRF_T + t];

  // Async copy of scores[l, b, :, :] into LDS buffer `buf`.
  // 4096 f32 = 1024 x b128; 128 threads -> 8 async-load instructions per wave.
  auto issue_tile = [&](int l, int buf) {
    const float* src = scores + ((size_t)l * CRF_B + b) * (CRF_T * CRF_T);
    unsigned lds_base = (unsigned)(uintptr_t)(&tile[buf][0]);
#pragma unroll
    for (int c = 0; c < 8; ++c) {
      int elt = (c * 128 + t) * 4;                       // float index
      unsigned lds_addr = lds_base + (unsigned)(elt * 4);
      unsigned long long gaddr = (unsigned long long)(uintptr_t)(src + elt);
      asm volatile("global_load_async_to_lds_b128 %0, %1, off"
                   :: "v"(lds_addr), "v"(gaddr) : "memory");
    }
  };

  issue_tile(1, 1);   // prefetch first recurrence tile

  for (int l = 1; l < CRF_L; ++l) {
    const int buf = l & 1;
    __syncthreads();                      // prev compute done; part[] ready
    if (l + 1 < CRF_L) {
      issue_tile(l + 1, (l + 1) & 1);     // prefetch next (its buffer is free)
      asm volatile("s_wait_asynccnt 0x8" ::: "memory");  // tile l arrived
    } else {
      asm volatile("s_wait_asynccnt 0x0" ::: "memory");
    }

    // Pmax over part[] (uniform broadcast reads, every thread redundantly).
    float pmax = -3.4e38f;
#pragma unroll 8
    for (int i = 0; i < CRF_T; ++i) pmax = fmaxf(pmax, part[i]);
    if (t < CRF_T) pexp[t] = __expf(part[t] - pmax);
    __syncthreads();                      // tile-l complete for ALL waves; pexp ready

    // Exp-domain matvec: r[j] = sum_i exp(scores[i,j]) * pexp[i], K = 64
    // split into two 16x16x32 f16 WMMAs. A rows replicated with pexp.
    v8f acc = {};
#pragma unroll
    for (int c = 0; c < 2; ++c) {
      const int i0 = c * 32;
      v16h av, bv;
#pragma unroll
      for (int e = 0; e < 16; ++e) {
        // A (16x32 f16): lanes 0-15 hold K 0..15; lanes 16-31 K 16..31
        int kA = i0 + e + (hi16 ? 16 : 0);
        av[e] = (_Float16)pexp[kA];
        // B (32x16 f16): lanes 0-15 hold K 0..15; lanes 16-31 K 16..31
        int kB = i0 + e + (hi16 ? 16 : 0);
        bv[e] = (_Float16)__expf(tile[buf][kB * CRF_T + j]);
      }
      acc = __builtin_amdgcn_wmma_f32_16x16x32_f16(
          /*neg_a=*/false, av, /*neg_b=*/false, bv,
          /*c_mod=*/(short)0, acc, /*reuse_a=*/false, /*reuse_b=*/false);
    }

    // D row M=0 lives in VGPR0 of lanes 0-15.
    const unsigned char m = mask[l * CRF_B + b];
    if (lane < 16 && m) part[j] = pmax + __logf(acc[0]);
    // part-write vs next-iteration read ordered by the loop-top barrier.
  }

  __syncthreads();
  if (t == 0) ws_part[b] = part[END_TAG];
}

// -------------------------------------------------------------- finalize ----
__global__ void crf_final_kernel(const float* __restrict__ ws_tg,
                                 const float* __restrict__ ws_part,
                                 float* __restrict__ out) {
  if (threadIdx.x == 0) {
    float tg = 0.0f, ps = 0.0f;
    for (int i = 0; i < CRF_B; ++i) tg += ws_tg[i];
    for (int i = 0; i < CRF_B; ++i) ps += ws_part[i];
    out[0] = (ps - tg) / (float)CRF_B;
  }
}

// ---------------------------------------------------------------- launch ----
extern "C" void kernel_launch(void* const* d_in, const int* in_sizes, int n_in,
                              void* d_out, int out_size, void* d_ws, size_t ws_size,
                              hipStream_t stream) {
  const float*         scores = (const float*)d_in[0];
  const long long*     target = (const long long*)d_in[1];   // int64
  const unsigned char* mask   = (const unsigned char*)d_in[2]; // bool (1 byte)
  float* out = (float*)d_out;

  float* wsf     = (float*)d_ws;
  float* ws_tg   = wsf;            // 128 block partials of tg_energy
  float* ws_part = wsf + CRF_B;    // 128 partition[b, END_TAG]

  // L*B = 32768 gathers -> 128 blocks x 256 threads
  crf_gather_kernel<<<128, 256, 0, stream>>>(scores, target, mask, ws_tg);
  // one workgroup per batch element
  crf_forward_kernel<<<CRF_B, 128, 0, stream>>>(scores, mask, ws_part);
  crf_final_kernel<<<1, 32, 0, stream>>>(ws_tg, ws_part, out);

  (void)in_sizes; (void)n_in; (void)out_size; (void)ws_size;
}
